// SimpleMixingHead_49632642072718
// MI455X (gfx1250) — matrix operand order repaired and checked by reading, compiled, not run on hardware
//
#include <hip/hip_runtime.h>

typedef __attribute__((ext_vector_type(16))) __bf16 v16bf;
typedef __attribute__((ext_vector_type(8)))  __bf16 v8bf;
typedef __attribute__((ext_vector_type(8)))  float  v8f;
typedef __attribute__((ext_vector_type(4)))  __bf16 v4bf;

#if defined(__gfx1250__)
#define HAVE_ASYNC 1
#else
#define HAVE_ASYNC 0
#endif

// 16B global -> LDS copy. On gfx1250: GLOBAL_LOAD_ASYNC_TO_LDS_B128
// (ASYNCcnt-tracked, memory pipe writes LDS directly). Fallback: sync copy.
// vdst = wave-relative LDS byte address = low 32 bits of the generic pointer.
__device__ __forceinline__ void cp_async16(void* lds, const void* gsrc) {
#if HAVE_ASYNC
    asm volatile("global_load_async_to_lds_b128 %0, %1, off"
                 :
                 : "v"((unsigned int)(unsigned long long)lds),
                   "v"((unsigned long long)gsrc)
                 : "memory");
#else
    *(uint4*)lds = *(const uint4*)gsrc;
#endif
}

#if HAVE_ASYNC
#define ASYNC_WAIT(n) asm volatile("s_wait_asynccnt " #n ::: "memory")
#else
#define ASYNC_WAIT(n) ((void)0)
#endif

__device__ __forceinline__ v16bf cat8(v8bf a, v8bf b) {
    return __builtin_shufflevector(a, b, 0, 1, 2, 3, 4, 5, 6, 7,
                                         8, 9, 10, 11, 12, 13, 14, 15);
}

// A/P-fragment (16x32 bf16, M x K): per lane two contiguous 8-elem runs at
// k = hi*8 and k = 16 + hi*8.  `row` points at row (m0 + lane%16), k0 added.
__device__ __forceinline__ v16bf load_afrag(const __bf16* row, int hi) {
    v8bf lo = *(const v8bf*)(row + hi * 8);
    v8bf hv = *(const v8bf*)(row + 16 + hi * 8);
    return cat8(lo, hv);
}
// B-fragment (32x16 bf16, K x N): lane n = lane%16 holds two 8-elem runs at
// k = hi*8 and k = 16 + hi*8 in the K-major source row.
__device__ __forceinline__ v16bf load_bfrag(const __bf16* row, int hi) {
    v8bf lo = *(const v8bf*)(row + hi * 8);
    v8bf hv = *(const v8bf*)(row + 16 + hi * 8);
    return cat8(lo, hv);
}

// ---------------------------------------------------------------------------
// f32 -> bf16 streaming conversion
// ---------------------------------------------------------------------------
__global__ __launch_bounds__(256) void cvt_bf16_kernel(const float* __restrict__ in,
                                                       __bf16* __restrict__ out, int n) {
    int i = (blockIdx.x * blockDim.x + threadIdx.x) * 4;
    if (i + 3 < n) {
        float4 f = *(const float4*)(in + i);
        v4bf v = { (__bf16)f.x, (__bf16)f.y, (__bf16)f.z, (__bf16)f.w };
        *(v4bf*)(out + i) = v;
    }
}

// ---------------------------------------------------------------------------
// Per-batch transpose: XT[b][n][s] = X[b][s][n]   (bf16, 64x64 LDS tiles)
// grid: (DM/64, T/64, B)
// ---------------------------------------------------------------------------
__global__ __launch_bounds__(256) void transpose_bf16(const __bf16* __restrict__ in,
                                                      __bf16* __restrict__ outp) {
    __shared__ __bf16 t[64][72];
    const int tid = threadIdx.x;
    const int b = blockIdx.z, n0 = blockIdx.x * 64, s0 = blockIdx.y * 64;
#pragma unroll
    for (int i = 0; i < 2; ++i) {
        int idx = tid + i * 256;
        int r = idx >> 3, c = (idx & 7) * 8;  // r: s-row, c: n-chunk
        *(v8bf*)&t[r][c] = *(const v8bf*)&in[((size_t)(b * 256 + s0 + r)) * 2048 + n0 + c];
    }
    __syncthreads();
#pragma unroll
    for (int i = 0; i < 2; ++i) {
        int idx = tid + i * 256;
        int r = idx >> 3, c = (idx & 7) * 8;  // r: n-row, c: s-chunk
        v8bf v = { t[c + 0][r], t[c + 1][r], t[c + 2][r], t[c + 3][r],
                   t[c + 4][r], t[c + 5][r], t[c + 6][r], t[c + 7][r] };
        *(v8bf*)&outp[((size_t)b * 2048 + n0 + r) * 256 + s0 + c] = v;
    }
}

// ---------------------------------------------------------------------------
// C[M,N](bf16) = A[M,K](bf16) * Bw[N,K](bf16)^T
// 128x64 tile, 8 waves x (32x32), async double-buffered K=32 LDS slabs.
// ---------------------------------------------------------------------------
__global__ __launch_bounds__(256) void gemm_bt_bf16(const __bf16* __restrict__ A,
                                                    const __bf16* __restrict__ Bw,
                                                    __bf16* __restrict__ C,
                                                    int M, int N, int K) {
    __shared__ __bf16 sA[2][128][40];
    __shared__ __bf16 sB[2][64][40];

    const int tid = threadIdx.x;
    const int wave = tid >> 5, lane = tid & 31;
    const int nlane = lane & 15, hi = lane >> 4;
    const int m0 = blockIdx.y * 128;
    const int n0 = blockIdx.x * 64;
    const int wm = (wave & 3) * 32;
    const int wn = (wave >> 2) * 32;

    auto prefetch = [&](int buf, int k0) {
#pragma unroll
        for (int i = 0; i < 2; ++i) {
            int idx = tid + i * 256;
            int r = idx >> 2, c = (idx & 3) * 8;
            cp_async16(&sA[buf][r][c], &A[(size_t)(m0 + r) * K + k0 + c]);
        }
        int r = tid >> 2, c = (tid & 3) * 8;
        cp_async16(&sB[buf][r][c], &Bw[(size_t)(n0 + r) * K + k0 + c]);
    };

    v8f acc[2][2] = {};
    const int nk = K >> 5;
    prefetch(0, 0);
#pragma unroll 1
    for (int ki = 0; ki < nk; ++ki) {
        const int buf = ki & 1;
        if (ki + 1 < nk) {
            prefetch(buf ^ 1, (ki + 1) * 32);
            ASYNC_WAIT(3);  // 3 outstanding = next slab; this slab complete
        } else {
            ASYNC_WAIT(0);
        }
        __syncthreads();

        v16bf af[2], bf[2];
#pragma unroll
        for (int t = 0; t < 2; ++t) {
            af[t] = load_afrag(&sA[buf][wm + t * 16 + nlane][0], hi);
            bf[t] = load_bfrag(&sB[buf][wn + t * 16 + nlane][0], hi);
        }
#pragma unroll
        for (int i = 0; i < 2; ++i)
#pragma unroll
            for (int j = 0; j < 2; ++j)
                acc[i][j] = __builtin_amdgcn_wmma_f32_16x16x32_bf16(
                    false, af[i], false, bf[j], (short)0, acc[i][j], false, false);
        __syncthreads();  // protect buf^1 (WAR) before next prefetch
    }

#pragma unroll
    for (int i = 0; i < 2; ++i)
#pragma unroll
        for (int j = 0; j < 2; ++j)
#pragma unroll
            for (int r = 0; r < 8; ++r) {
                int m = m0 + wm + i * 16 + hi * 8 + r;
                int n = n0 + wn + j * 16 + nlane;
                C[(size_t)m * N + n] = (__bf16)acc[i][j][r];
            }
}

// ---------------------------------------------------------------------------
// Attention per (batch, 64-row query tile):
//   S = Q X^T * scale (causal) -> softmax -> O = P X (via XT)
// Dynamic LDS: [0,64KB) f32 scores S[64][256]; [64KB, +40KB) X slab x2 buffers.
// ---------------------------------------------------------------------------
__global__ __launch_bounds__(256) void attn_kernel(const __bf16* __restrict__ Qb,
                                                   const __bf16* __restrict__ Xb,
                                                   const __bf16* __restrict__ XT,
                                                   float* __restrict__ out) {
    extern __shared__ char smem[];
    float (*S)[256] = (float (*)[256])smem;
    __bf16 (*sX)[256][40] = (__bf16 (*)[256][40])(smem + 65536);

    const int b  = blockIdx.y;
    const int t0 = blockIdx.x * 64;
    const int tid = threadIdx.x;
    const int wave = tid >> 5, lane = tid & 31;
    const int nlane = lane & 15, hi = lane >> 4;

    const size_t baseQ = ((size_t)b * 256 + t0) * 2048;
    const size_t baseX = (size_t)b * 256 * 2048;
    const size_t baseT = (size_t)b * 2048 * 256;
    const float scale = 0.022097086912079608f;  // 1/sqrt(2048)

    const int mt = wave & 3;    // query row tile (16 rows)
    const int grp = wave >> 2;  // 0 or 1
    const int sg = grp * 8;     // key tile group for phase 1

    auto prefx = [&](int buf, int k0) {
#pragma unroll
        for (int i = 0; i < 4; ++i) {
            int idx = tid + i * 256;
            int r = idx >> 2, c = (idx & 3) * 8;  // r: token, c: k-chunk
            cp_async16(&sX[buf][r][c], &Xb[baseX + (size_t)r * 2048 + k0 + c]);
        }
    };

    // ---------------- Phase 1: S = Q X^T ----------------
    {
        v8f acc[8] = {};
        prefx(0, 0);
#pragma unroll 1
        for (int ki = 0; ki < 64; ++ki) {
            const int buf = ki & 1;
            const int k0 = ki * 32;
            if (ki + 1 < 64) {
                prefx(buf ^ 1, (ki + 1) * 32);
                ASYNC_WAIT(4);
            } else {
                ASYNC_WAIT(0);
            }
            __syncthreads();

            v16bf af = load_afrag(&Qb[baseQ + (size_t)(mt * 16 + nlane) * 2048 + k0], hi);
#pragma unroll
            for (int st = 0; st < 8; ++st) {
                v16bf bf = load_bfrag(&sX[buf][(sg + st) * 16 + nlane][0], hi);
                acc[st] = __builtin_amdgcn_wmma_f32_16x16x32_bf16(
                    false, af, false, bf, (short)0, acc[st], false, false);
            }
            __syncthreads();
        }
#pragma unroll
        for (int st = 0; st < 8; ++st) {
            const int stile = sg + st;
#pragma unroll
            for (int r = 0; r < 8; ++r) {
                int trow = mt * 16 + hi * 8 + r;  // local query row
                int scol = stile * 16 + nlane;    // key index
                float v = acc[st][r] * scale;
                if (scol > t0 + trow) v = -__builtin_inff();
                S[trow][scol] = v;
            }
        }
    }
    __syncthreads();

    // ---------------- Phase 2: softmax; pack P(bf16) in-place per row -----
    if (tid < 64) {
        const int row = tid;
        float mx = -__builtin_inff();
#pragma unroll 1
        for (int i = 0; i < 256; ++i) mx = fmaxf(mx, S[row][i]);
        float sum = 0.f;
#pragma unroll 1
        for (int i = 0; i < 256; ++i) {
            float e = __expf(S[row][i] - mx);
            S[row][i] = e;
            sum += e;
        }
        const float inv = 1.0f / sum;
        __bf16* Prow = (__bf16*)&S[row][0];  // own row only: no aliasing
#pragma unroll 1
        for (int i = 0; i < 256; ++i) Prow[i] = (__bf16)(S[row][i] * inv);
    }
    __syncthreads();

    // ---------------- Phase 3: O = P X  (B-frags from XT, contiguous) -----
    {
        const __bf16* P = (const __bf16*)smem;  // row stride 512 bf16
        const __bf16* Pr = P + (size_t)(mt * 16 + nlane) * 512;
        v16bf pf[8];
#pragma unroll
        for (int ks = 0; ks < 8; ++ks)
            pf[ks] = load_afrag(Pr + ks * 32, hi);

#pragma unroll 1
        for (int j = 0; j < 64; ++j) {
            const int n0 = (grp + 2 * j) * 16;  // DM column tile
            const __bf16* xtr = &XT[baseT + (size_t)(n0 + nlane) * 256];
            v8f o = {};
#pragma unroll
            for (int ks = 0; ks < 8; ++ks) {
                v16bf bf = load_bfrag(xtr + ks * 32, hi);
                o = __builtin_amdgcn_wmma_f32_16x16x32_bf16(
                    false, pf[ks], false, bf, (short)0, o, false, false);
            }
#pragma unroll
            for (int r = 0; r < 8; ++r)
                out[((size_t)b * 256 + t0 + mt * 16 + hi * 8 + r) * 2048 + n0 + nlane] = o[r];
        }
    }
}

// ---------------------------------------------------------------------------
// Workspace (bytes):
//   Xb   @ 0          : 67108864   (16384x2048 bf16)
//   Wqb  @ 67108864   : 1048576
//   Wktb @ 68157440   : 1048576
//   T1b  @ 69206016   : 8388608    (16384x256 bf16)
//   Qb   @ 77594624   : 67108864
//   XTb  @ 144703488  : 67108864   (per-batch X^T)   total ~202 MB
// ---------------------------------------------------------------------------
extern "C" void kernel_launch(void* const* d_in, const int* in_sizes, int n_in,
                              void* d_out, int out_size, void* d_ws, size_t ws_size,
                              hipStream_t stream) {
    const float* x   = (const float*)d_in[0];
    const float* Wq  = (const float*)d_in[1];
    const float* Wkt = (const float*)d_in[2];
    float* out = (float*)d_out;

    char* ws = (char*)d_ws;
    __bf16* Xb   = (__bf16*)(ws);
    __bf16* Wqb  = (__bf16*)(ws + 67108864u);
    __bf16* Wktb = (__bf16*)(ws + 68157440u);
    __bf16* T1b  = (__bf16*)(ws + 69206016u);
    __bf16* Qb   = (__bf16*)(ws + 77594624u);
    __bf16* XTb  = (__bf16*)(ws + 144703488u);

    cvt_bf16_kernel<<<32768, 256, 0, stream>>>(x,   Xb,   64 * 256 * 2048);
    cvt_bf16_kernel<<<512,   256, 0, stream>>>(Wq,  Wqb,  256 * 2048);
    cvt_bf16_kernel<<<512,   256, 0, stream>>>(Wkt, Wktb, 2048 * 256);

    transpose_bf16<<<dim3(32, 4, 64), 256, 0, stream>>>(Xb, XTb);

    // T1[16384,256] = Xb * Wqb^T ; Qb[16384,2048] = T1b * Wktb^T
    gemm_bt_bf16<<<dim3(4, 128), 256, 0, stream>>>(Xb, Wqb, T1b, 16384, 256, 2048);
    gemm_bt_bf16<<<dim3(32, 128), 256, 0, stream>>>(T1b, Wktb, Qb, 16384, 2048, 256);

    attn_kernel<<<dim3(4, 64), 256, 106496, stream>>>(Qb, Xb, XTb, out);
}
